// GlobalInteractorLayer_80942953661181
// MI455X (gfx1250) — matrix operand order repaired and checked by reading, not compiled, not run on hardware
//
#include <hip/hip_runtime.h>

// ---- CDNA5 / gfx1250: wave32, WMMA bf16 16x16x32, NT loads for streamed pe ----

typedef __attribute__((ext_vector_type(16))) __bf16 v16bf;
typedef __attribute__((ext_vector_type(8)))  __bf16 bf16x8;
typedef __attribute__((ext_vector_type(8)))  float   v8f;
typedef __attribute__((ext_vector_type(4)))  float   f4;

#define EMBED 128
#define HEADS 8
#define DHEAD 16
#define FFN_H 512
#define BB    16
#define NN    256
#define ROWS  (BB * NN)   // 4096

static __device__ __forceinline__ __bf16 f2bf(float f) {
  union { float f; unsigned u; } x; x.f = f;
  unsigned r = (x.u + 0x7FFFu + ((x.u >> 16) & 1u)) >> 16;   // round-to-nearest-even
  union { unsigned short s; __bf16 b; } y; y.s = (unsigned short)r;
  return y.b;
}

static __device__ __forceinline__ float wave_sum(float v) {
#pragma unroll
  for (int off = 16; off > 0; off >>= 1) v += __shfl_xor(v, off, 32);
  return v;
}
static __device__ __forceinline__ float wave_max(float v) {
#pragma unroll
  for (int off = 16; off > 0; off >>= 1) v = fmaxf(v, __shfl_xor(v, off, 32));
  return v;
}

// ---------------------------------------------------------------------------
// Weight prep: out[n*K + k] = (bf16) in[k*N + n]   (transpose so GEMM B-operand
// reads contiguous K per lane)
// ---------------------------------------------------------------------------
__global__ void transpose_f32_to_bf16(const float* __restrict__ in,
                                      __bf16* __restrict__ out, int K, int N) {
  int idx = blockIdx.x * 256 + threadIdx.x;
  if (idx >= K * N) return;
  int n = idx / K, kk = idx % K;
  out[idx] = f2bf(in[kk * N + n]);
}

// ---------------------------------------------------------------------------
// LayerNorm over EMBED=128, one wave per row (32 lanes x 4 floats), bf16 out.
// ---------------------------------------------------------------------------
__global__ void layernorm_bf16(const float* __restrict__ x,
                               const float* __restrict__ g,
                               const float* __restrict__ bvec,
                               __bf16* __restrict__ out) {
  int wave = threadIdx.x >> 5, lane = threadIdx.x & 31;
  int row = blockIdx.x * 8 + wave;
  const f4* xr = (const f4*)(x + (size_t)row * EMBED);
  f4 v = xr[lane];
  float s  = v.x + v.y + v.z + v.w;
  float sq = v.x * v.x + v.y * v.y + v.z * v.z + v.w * v.w;
  s = wave_sum(s); sq = wave_sum(sq);
  float mu  = s * (1.0f / EMBED);
  float var = sq * (1.0f / EMBED) - mu * mu;
  float rs  = rsqrtf(var + 1e-5f);
  f4 gv = ((const f4*)g)[lane];
  f4 bv = ((const f4*)bvec)[lane];
  __bf16* op = out + (size_t)row * EMBED + lane * 4;
  op[0] = f2bf((v.x - mu) * rs * gv.x + bv.x);
  op[1] = f2bf((v.y - mu) * rs * gv.y + bv.y);
  op[2] = f2bf((v.z - mu) * rs * gv.z + bv.z);
  op[3] = f2bf((v.w - mu) * rs * gv.w + bv.w);
}

// ---------------------------------------------------------------------------
// bf16 WMMA GEMM: C[M,N] = A[M,K] @ B[K,N] (+bias, optional ReLU, +residual).
// Bt is B stored transposed [N,K] bf16. One wave per 16x16 output tile,
// 4 waves per block (distinct n-tiles). K multiple of 32; M,N multiples of 16.
//
// Register layouts used here (A/B K order is contiguous per half-wave, not the ISA table order):
//  A (16x32 bf16): lanes 0-15 row m=lane, K in {0..15}; lanes 16-31
//                  same rows, K in {16..31}.
//  B (32x16 bf16): lane -> col n=lane&15; half-wave selects K base 0 / 16.
//  D (16x16 f32):  vgpr r -> M = r + 8*(lane>=16), N = lane&15.
// ---------------------------------------------------------------------------
__global__ void gemm_bf16_wmma(const __bf16* __restrict__ A,
                               const __bf16* __restrict__ Bt,
                               const float*  __restrict__ bias,
                               const float*  __restrict__ res,
                               float*        __restrict__ Cf,
                               __bf16*       __restrict__ Cb,
                               int N, int K, int relu) {
  int lane = threadIdx.x & 31;
  int wave = threadIdx.x >> 5;
  int m0 = blockIdx.x * 16;
  int n0 = (blockIdx.y * 4 + wave) * 16;
  int half = lane >> 4;        // 0: lanes 0-15, 1: lanes 16-31
  int lr   = lane & 15;

  const __bf16* ap = A  + (size_t)(m0 + lr) * K + half * 16;
  const __bf16* bp = Bt + (size_t)(n0 + lr) * K + half * 16;

  v8f acc = {};
  for (int k0 = 0; k0 < K; k0 += 32) {
    bf16x8 alo = *(const bf16x8*)(ap + k0);        // K = k0 + half*16 + 0..7
    bf16x8 ahi = *(const bf16x8*)(ap + k0 + 8);    // K = k0 + half*16 + 8..15
    v16bf a = __builtin_shufflevector(alo, ahi, 0, 1, 2, 3, 4, 5, 6, 7,
                                                8, 9, 10, 11, 12, 13, 14, 15);
    v16bf b = *(const v16bf*)(bp + k0);            // K = k0 + half*16 + 0..15
    acc = __builtin_amdgcn_wmma_f32_16x16x32_bf16(
        /*neg_a=*/false, a, /*neg_b=*/false, b,
        /*c_mod=*/(short)0, acc, /*reuse_a=*/false, /*reuse_b=*/false);
  }

  int ncol = n0 + lr;
  float bscal = bias ? bias[ncol] : 0.0f;
#pragma unroll
  for (int r = 0; r < 8; ++r) {
    int mrow = m0 + r + half * 8;
    float val = acc[r] + bscal;
    if (relu) val = fmaxf(val, 0.0f);
    size_t oi = (size_t)mrow * N + ncol;
    if (res) val += res[oi];
    if (Cf) Cf[oi] = val;
    if (Cb) Cb[oi] = f2bf(val);
  }
}

// ---------------------------------------------------------------------------
// Fused attention: scores = (q.k + q.pe)/sqrt(16) + mask ; softmax ; ctx = p@v.
// Block = (b,i) pair, wave = head. Each lane owns 8 j's. pe (512MB, used once)
// is read with non-temporal hints; k/v (2MB each) stay L2-resident.
// ---------------------------------------------------------------------------
__global__ void attention_fused(const float* __restrict__ q,
                                const float* __restrict__ k,
                                const float* __restrict__ v,
                                const float* __restrict__ pe,
                                const float* __restrict__ mask,
                                __bf16* __restrict__ ctx) {
  int i = blockIdx.x, b = blockIdx.y;
  int h = threadIdx.x >> 5, lane = threadIdx.x & 31;
  int rowqi = b * NN + i;

  const f4* qp = (const f4*)(q + (size_t)rowqi * EMBED + h * DHEAD);
  f4 q0 = qp[0], q1 = qp[1], q2 = qp[2], q3 = qp[3];

  const float* kb  = k  + (size_t)b * NN * EMBED + h * DHEAD;
  const float* vb  = v  + (size_t)b * NN * EMBED + h * DHEAD;
  const float* peb = pe + (size_t)rowqi * NN * EMBED + h * DHEAD;
  const float* mb  = mask + (size_t)rowqi * NN;

  float s[8];
#pragma unroll
  for (int t = 0; t < 8; ++t) {
    int j = lane + 32 * t;
    const f4* kr = (const f4*)(kb + (size_t)j * EMBED);
    const f4* pr = (const f4*)(peb + (size_t)j * EMBED);
    f4 k0 = kr[0], k1 = kr[1], k2 = kr[2], k3 = kr[3];
    f4 p0 = __builtin_nontemporal_load(pr + 0);
    f4 p1 = __builtin_nontemporal_load(pr + 1);
    f4 p2 = __builtin_nontemporal_load(pr + 2);
    f4 p3 = __builtin_nontemporal_load(pr + 3);
    float acc;
    acc  = q0.x * (k0.x + p0.x) + q0.y * (k0.y + p0.y) +
           q0.z * (k0.z + p0.z) + q0.w * (k0.w + p0.w);
    acc += q1.x * (k1.x + p1.x) + q1.y * (k1.y + p1.y) +
           q1.z * (k1.z + p1.z) + q1.w * (k1.w + p1.w);
    acc += q2.x * (k2.x + p2.x) + q2.y * (k2.y + p2.y) +
           q2.z * (k2.z + p2.z) + q2.w * (k2.w + p2.w);
    acc += q3.x * (k3.x + p3.x) + q3.y * (k3.y + p3.y) +
           q3.z * (k3.z + p3.z) + q3.w * (k3.w + p3.w);
    s[t] = acc * 0.25f + mb[j];           // 1/sqrt(DH) = 0.25
  }

  float m = s[0];
#pragma unroll
  for (int t = 1; t < 8; ++t) m = fmaxf(m, s[t]);
  m = wave_max(m);
  float sum = 0.0f;
#pragma unroll
  for (int t = 0; t < 8; ++t) { s[t] = __expf(s[t] - m); sum += s[t]; }
  sum = wave_sum(sum);
  float inv = 1.0f / sum;

  f4 c0 = {0.f, 0.f, 0.f, 0.f}, c1 = c0, c2 = c0, c3 = c0;
#pragma unroll
  for (int t = 0; t < 8; ++t) {
    int j = lane + 32 * t;
    const f4* vr = (const f4*)(vb + (size_t)j * EMBED);
    c0 += s[t] * vr[0];
    c1 += s[t] * vr[1];
    c2 += s[t] * vr[2];
    c3 += s[t] * vr[3];
  }
  c0.x = wave_sum(c0.x); c0.y = wave_sum(c0.y); c0.z = wave_sum(c0.z); c0.w = wave_sum(c0.w);
  c1.x = wave_sum(c1.x); c1.y = wave_sum(c1.y); c1.z = wave_sum(c1.z); c1.w = wave_sum(c1.w);
  c2.x = wave_sum(c2.x); c2.y = wave_sum(c2.y); c2.z = wave_sum(c2.z); c2.w = wave_sum(c2.w);
  c3.x = wave_sum(c3.x); c3.y = wave_sum(c3.y); c3.z = wave_sum(c3.z); c3.w = wave_sum(c3.w);

  if (lane == 0) {
    __bf16* cp = ctx + (size_t)rowqi * EMBED + h * DHEAD;
    cp[0]  = f2bf(c0.x * inv); cp[1]  = f2bf(c0.y * inv);
    cp[2]  = f2bf(c0.z * inv); cp[3]  = f2bf(c0.w * inv);
    cp[4]  = f2bf(c1.x * inv); cp[5]  = f2bf(c1.y * inv);
    cp[6]  = f2bf(c1.z * inv); cp[7]  = f2bf(c1.w * inv);
    cp[8]  = f2bf(c2.x * inv); cp[9]  = f2bf(c2.y * inv);
    cp[10] = f2bf(c2.z * inv); cp[11] = f2bf(c2.w * inv);
    cp[12] = f2bf(c3.x * inv); cp[13] = f2bf(c3.y * inv);
    cp[14] = f2bf(c3.z * inv); cp[15] = f2bf(c3.w * inv);
  }
}

// ---------------------------------------------------------------------------
extern "C" void kernel_launch(void* const* d_in, const int* in_sizes, int n_in,
                              void* d_out, int out_size, void* d_ws, size_t ws_size,
                              hipStream_t stream) {
  const float* x    = (const float*)d_in[0];
  const float* pe   = (const float*)d_in[1];
  const float* mask = (const float*)d_in[2];
  const float* ln1g = (const float*)d_in[3];
  const float* ln1b = (const float*)d_in[4];
  const float* Wq   = (const float*)d_in[5];
  const float* bq   = (const float*)d_in[6];
  const float* Wk   = (const float*)d_in[7];
  const float* bk   = (const float*)d_in[8];
  const float* Wv   = (const float*)d_in[9];
  const float* bvv  = (const float*)d_in[10];
  const float* Wo   = (const float*)d_in[11];
  const float* bo   = (const float*)d_in[12];
  const float* ln2g = (const float*)d_in[13];
  const float* ln2b = (const float*)d_in[14];
  const float* W1   = (const float*)d_in[15];
  const float* b1   = (const float*)d_in[16];
  const float* W2   = (const float*)d_in[17];
  const float* b2   = (const float*)d_in[18];
  float* out = (float*)d_out;

  // workspace carve-out (~15.5 MB total)
  char* ws = (char*)d_ws;
  size_t off = 0;
  auto carve = [&](size_t bytes) -> char* {
    char* p = ws + off;
    off += (bytes + 255) & ~(size_t)255;
    return p;
  };
  __bf16* WqT = (__bf16*)carve((size_t)EMBED * EMBED * 2);
  __bf16* WkT = (__bf16*)carve((size_t)EMBED * EMBED * 2);
  __bf16* WvT = (__bf16*)carve((size_t)EMBED * EMBED * 2);
  __bf16* WoT = (__bf16*)carve((size_t)EMBED * EMBED * 2);
  __bf16* W1T = (__bf16*)carve((size_t)EMBED * FFN_H * 2);
  __bf16* W2T = (__bf16*)carve((size_t)FFN_H * EMBED * 2);
  __bf16* xn1 = (__bf16*)carve((size_t)ROWS * EMBED * 2);
  float*  qf  = (float*)carve((size_t)ROWS * EMBED * 4);
  float*  kf  = (float*)carve((size_t)ROWS * EMBED * 4);
  float*  vf  = (float*)carve((size_t)ROWS * EMBED * 4);
  __bf16* ctx = (__bf16*)carve((size_t)ROWS * EMBED * 2);
  float*  x2  = (float*)carve((size_t)ROWS * EMBED * 4);
  __bf16* xn2 = (__bf16*)carve((size_t)ROWS * EMBED * 2);
  __bf16* hid = (__bf16*)carve((size_t)ROWS * FFN_H * 2);

  // 0) weights -> transposed bf16 (tiny, L2-resident thereafter)
  transpose_f32_to_bf16<<<(EMBED * EMBED + 255) / 256, 256, 0, stream>>>(Wq, WqT, EMBED, EMBED);
  transpose_f32_to_bf16<<<(EMBED * EMBED + 255) / 256, 256, 0, stream>>>(Wk, WkT, EMBED, EMBED);
  transpose_f32_to_bf16<<<(EMBED * EMBED + 255) / 256, 256, 0, stream>>>(Wv, WvT, EMBED, EMBED);
  transpose_f32_to_bf16<<<(EMBED * EMBED + 255) / 256, 256, 0, stream>>>(Wo, WoT, EMBED, EMBED);
  transpose_f32_to_bf16<<<(EMBED * FFN_H + 255) / 256, 256, 0, stream>>>(W1, W1T, EMBED, FFN_H);
  transpose_f32_to_bf16<<<(FFN_H * EMBED + 255) / 256, 256, 0, stream>>>(W2, W2T, FFN_H, EMBED);

  // 1) LN1 -> bf16
  layernorm_bf16<<<ROWS / 8, 256, 0, stream>>>(x, ln1g, ln1b, xn1);

  // 2) QKV projections (WMMA bf16, f32 accumulate)
  dim3 g128(ROWS / 16, EMBED / 64);   // (256, 2)
  gemm_bf16_wmma<<<g128, 128, 0, stream>>>(xn1, WqT, bq,  nullptr, qf, nullptr, EMBED, EMBED, 0);
  gemm_bf16_wmma<<<g128, 128, 0, stream>>>(xn1, WkT, bk,  nullptr, kf, nullptr, EMBED, EMBED, 0);
  gemm_bf16_wmma<<<g128, 128, 0, stream>>>(xn1, WvT, bvv, nullptr, vf, nullptr, EMBED, EMBED, 0);

  // 3) fused rel-pos attention (streams 512MB pe with NT hints; bandwidth-bound)
  attention_fused<<<dim3(NN, BB), 256, 0, stream>>>(qf, kf, vf, pe, mask, ctx);

  // 4) output projection + residual -> x2
  gemm_bf16_wmma<<<g128, 128, 0, stream>>>(ctx, WoT, bo, x, x2, nullptr, EMBED, EMBED, 0);

  // 5) LN2 -> bf16
  layernorm_bf16<<<ROWS / 8, 256, 0, stream>>>(x2, ln2g, ln2b, xn2);

  // 6) FFN: relu(xn2 @ W1 + b1) -> bf16 hidden ; hidden @ W2 + b2 + x2 -> out
  dim3 g512(ROWS / 16, FFN_H / 64);   // (256, 8)
  gemm_bf16_wmma<<<g512, 128, 0, stream>>>(xn2, W1T, b1, nullptr, nullptr, hid, FFN_H, EMBED, 1);
  gemm_bf16_wmma<<<g128, 128, 0, stream>>>(hid, W2T, b2, x2, out, nullptr, EMBED, FFN_H, 0);
}
